// ArcGenerator_3195455668470
// MI455X (gfx1250) — matrix operand order repaired and checked by reading, not compiled, not run on hardware
//
#include <hip/hip_runtime.h>
#include <hip/hip_bf16.h>
#include <math.h>

typedef _Float16 f16;
typedef __attribute__((ext_vector_type(4)))  _Float16 v4h;
typedef __attribute__((ext_vector_type(8)))  _Float16 v8h;
typedef __attribute__((ext_vector_type(16))) _Float16 v16h;
typedef __attribute__((ext_vector_type(8)))  float    v8f;
typedef __attribute__((ext_vector_type(4)))  int      v4i;

static constexpr int kT = 1024, kS = 1024, kB = 8, kE = 768, kH = 12, kD = 64;

#define GLOBAL_AS __attribute__((address_space(1)))
#define LDS_AS    __attribute__((address_space(3)))

__device__ __forceinline__ v8f wmma_f16(v16h a, v16h b, v8f c) {
  return __builtin_amdgcn_wmma_f32_16x16x32_f16(false, a, false, b, (short)0, c,
                                                false, false);
}
__device__ __forceinline__ v16h cat8(v8h lo, v8h hi) {
  return __builtin_shufflevector(lo, hi, 0, 1, 2, 3, 4, 5, 6, 7,
                                 8, 9, 10, 11, 12, 13, 14, 15);
}
__device__ __forceinline__ v4h cvt4(float4 f) {
  v4h r; r[0] = (f16)f.x; r[1] = (f16)f.y; r[2] = (f16)f.z; r[3] = (f16)f.w;
  return r;
}

#if __has_builtin(__builtin_amdgcn_global_load_async_to_lds_b128)
#define HAS_ASYNC_LDS 1
#else
#define HAS_ASYNC_LDS 0
#endif

__device__ __forceinline__ void cp16_async(f16* lds, const f16* g) {
#if HAS_ASYNC_LDS
  __builtin_amdgcn_global_load_async_to_lds_b128(
      (GLOBAL_AS v4i*)g, (LDS_AS v4i*)lds, 0, 0);
#else
  *(uint4*)lds = *(const uint4*)g;
#endif
}
__device__ __forceinline__ void wait_async0() {
#if HAS_ASYNC_LDS && __has_builtin(__builtin_amdgcn_s_wait_asynccnt)
  __builtin_amdgcn_s_wait_asynccnt(0);
#endif
}

__device__ __forceinline__ float gmax16(float v) {
  v = fmaxf(v, __shfl_xor(v, 1));
  v = fmaxf(v, __shfl_xor(v, 2));
  v = fmaxf(v, __shfl_xor(v, 4));
  v = fmaxf(v, __shfl_xor(v, 8));
  return v;
}
__device__ __forceinline__ float gsum16(float v) {
  v += __shfl_xor(v, 1);
  v += __shfl_xor(v, 2);
  v += __shfl_xor(v, 4);
  v += __shfl_xor(v, 8);
  return v;
}

// ---------------------------------------------------------------------------
// Kernel 1: projection (templated: 0=Q scaled, 1=K, 2=V transposed [b,h,d,s]).
// ---------------------------------------------------------------------------
template <int WHICH>
__global__ void proj_kernel(const float* __restrict__ X,
                            const float* __restrict__ Win,
                            const float* __restrict__ b_in,
                            f16* __restrict__ dst) {
  const float* W = Win + (size_t)WHICH * kE * kE;
  const int row0 = blockIdx.x * 16, col0 = blockIdx.y * 64;
  __shared__ alignas(32) f16 As[16][32];
  __shared__ alignas(32) f16 Bs[64][32];
  __shared__ alignas(16) f16 Cs[16][64];
  const int lane = threadIdx.x & 31, g = lane >> 4, m = lane & 15;
  v8f c[4] = {};
  for (int k0 = 0; k0 < kE; k0 += 32) {
    if (k0 + 32 < kE) {
      __builtin_prefetch(X + (size_t)(row0 + m) * kE + k0 + 32, 0, 1);
      __builtin_prefetch(W + (size_t)(col0 + 2 * lane) * kE + k0 + 32, 0, 1);
    }
#pragma unroll
    for (int j = 0; j < 4; ++j) {                       // A: 16x32 f32 -> f16
      int i = j * 32 + lane, r = i >> 3, q = i & 7;
      *(v4h*)&As[r][q * 4] = cvt4(((const float4*)(X + (size_t)(row0 + r) * kE + k0))[q]);
    }
#pragma unroll
    for (int j = 0; j < 16; ++j) {                      // B^T: 64x32 f32 -> f16
      int i = j * 32 + lane, n = i >> 3, q = i & 7;
      *(v4h*)&Bs[n][q * 4] = cvt4(((const float4*)(W + (size_t)(col0 + n) * kE + k0))[q]);
    }
    __syncthreads();
    v16h a = cat8(*(const v8h*)&As[m][g * 8], *(const v8h*)&As[m][16 + g * 8]);
#pragma unroll
    for (int nt = 0; nt < 4; ++nt) {
      v16h bv = cat8(*(const v8h*)&Bs[nt * 16 + m][g * 8], *(const v8h*)&Bs[nt * 16 + m][16 + g * 8]);
      c[nt] = wmma_f16(a, bv, c[nt]);
    }
    __syncthreads();
  }
  const float scale = (WHICH == 0) ? 0.125f : 1.0f;  // D^-0.5
#pragma unroll
  for (int nt = 0; nt < 4; ++nt) {
    float bias = b_in[(size_t)WHICH * kE + col0 + nt * 16 + m];
#pragma unroll
    for (int r = 0; r < 8; ++r)
      Cs[8 * g + r][nt * 16 + m] = (f16)((c[nt][r] + bias) * scale);
  }
  __syncthreads();
  const int h = col0 >> 6;
  if (WHICH != 2) {
#pragma unroll
    for (int j = 0; j < 4; ++j) {
      int i = j * 32 + lane, r = i >> 3, q = i & 7;
      int row = row0 + r, t = row >> 3, bb = row & 7;
      *(uint4*)(dst + (((size_t)bb * kH + h) * kT + t) * kD + q * 8) = ((uint4*)Cs)[i];
    }
  } else {
    const int t0 = row0 >> 3;  // even
#pragma unroll
    for (int j = 0; j < 16; ++j) {
      int i = j * 32 + lane, d = i >> 3, bb = i & 7;
      union { f16 h2[2]; unsigned u; } pk;
      pk.h2[0] = Cs[bb][d];
      pk.h2[1] = Cs[bb + 8][d];
      *(unsigned*)(dst + (((size_t)bb * kH + h) * kD + d) * kS + t0) = pk.u;
    }
  }
}

// ---------------------------------------------------------------------------
// Kernel 2: single-pass flash attention (online softmax + O accumulation).
// Double-buffered async K/V^T tiles (runtime buffer index: avoids spills).
// ---------------------------------------------------------------------------
__global__ void flash_attn_kernel(const f16* __restrict__ q_h,
                                  const f16* __restrict__ k_h,
                                  const f16* __restrict__ v_t,
                                  const float* __restrict__ amask,
                                  const unsigned char* __restrict__ pad,
                                  f16* __restrict__ attn_h,
                                  float* __restrict__ rmax_g,
                                  float* __restrict__ rsum_g) {
  const int t0 = blockIdx.x * 16, h = blockIdx.y, b = blockIdx.z;
  __shared__ alignas(32) f16 Qs[16][kD];
  __shared__ alignas(32) f16 Ks[2][16][kD];
  __shared__ alignas(32) f16 VsT[2][kD][16];
  __shared__ alignas(32) f16 Wt[16][16];
  __shared__ alignas(16) f16 Cs[16][64];
  const int lane = threadIdx.x & 31, g = lane >> 4, m = lane & 15;
  const f16* kplane = k_h + ((size_t)b * kH + h) * kS * kD;
  const f16* vplane = v_t + ((size_t)b * kH + h) * (size_t)kD * kS;
  const uint4* qsrc = (const uint4*)(q_h + (((size_t)b * kH + h) * kT + t0) * kD);
#pragma unroll
  for (int j = 0; j < 4; ++j) ((uint4*)Qs)[j * 32 + lane] = qsrc[j * 32 + lane];
  auto issue = [&](int buf, int s0) {
#pragma unroll
    for (int j = 0; j < 4; ++j) {
      int i = j * 32 + lane;
      cp16_async(&Ks[buf][0][0] + i * 8, kplane + (size_t)s0 * kD + i * 8);
    }
#pragma unroll
    for (int j = 0; j < 4; ++j) {
      int i = j * 32 + lane, d = i >> 1, q = i & 1;
      cp16_async(&VsT[buf][d][q * 8], vplane + (size_t)d * kS + s0 + q * 8);
    }
  };
  issue(0, 0);
  __syncthreads();
  v8f o[4] = {};
  float rm[8], rs[8];
#pragma unroll
  for (int r = 0; r < 8; ++r) { rm[r] = -INFINITY; rs[r] = 0.f; }
  for (int it = 0; it < kS / 16; ++it) {
    const int s0 = it * 16, cur = it & 1;
    wait_async0();
    __syncthreads();
    if (s0 + 16 < kS) issue(cur ^ 1, s0 + 16);
    v8f c = {};
#pragma unroll
    for (int kc = 0; kc < kD; kc += 32) {
      v16h a = cat8(*(const v8h*)&Qs[m][kc + g * 8],
                    *(const v8h*)&Qs[m][kc + 16 + g * 8]);
      v16h bv = cat8(*(const v8h*)&Ks[cur][m][kc + g * 8], *(const v8h*)&Ks[cur][m][kc + 16 + g * 8]);
      c = wmma_f16(a, bv, c);
    }
    const bool pmask = pad[(size_t)b * kS + s0 + m] != 0;
#pragma unroll
    for (int r = 0; r < 8; ++r) {
      int M = 8 * g + r;
      float l = c[r] + amask[(size_t)(t0 + M) * kS + s0 + m];
      if (pmask) l = -INFINITY;
      float tm = gmax16(l);
      float nm = fmaxf(rm[r], tm);
      float sc = (rm[r] == -INFINITY) ? 0.f : __expf(rm[r] - nm);
      float e  = (nm == -INFINITY) ? 0.f : __expf(l - nm);
      rs[r] = rs[r] * sc + gsum16(e);
      rm[r] = nm;
      Wt[M][m] = (f16)e;
#pragma unroll
      for (int nt = 0; nt < 4; ++nt) o[nt][r] *= sc;   // rescale O row
    }
    __syncthreads();
    v8h z8 = {};
    v16h a = cat8(*(const v8h*)&Wt[m][g * 8], z8);     // K padded 16->32
#pragma unroll
    for (int nt = 0; nt < 4; ++nt) {
      v16h bv = cat8(*(const v8h*)&VsT[cur][nt * 16 + m][g * 8], z8);
      o[nt] = wmma_f16(a, bv, o[nt]);
    }
    __syncthreads();
  }
#pragma unroll
  for (int nt = 0; nt < 4; ++nt)
#pragma unroll
    for (int r = 0; r < 8; ++r)
      Cs[8 * g + r][nt * 16 + m] = (f16)(o[nt][r] / rs[r]);
  __syncthreads();
#pragma unroll
  for (int j = 0; j < 4; ++j) {
    int i = j * 32 + lane, r = i >> 3, q = i & 7;
    *(uint4*)(attn_h + ((size_t)(t0 + r) * kB + b) * kE + (size_t)h * kD + q * 8) =
        ((uint4*)Cs)[i];
  }
  if (m == 0) {                                        // final row stats for BCE
    size_t base = ((size_t)b * kH + h) * kT + t0 + 8 * g;
#pragma unroll
    for (int r = 0; r < 8; ++r) { rmax_g[base + r] = rm[r]; rsum_g[base + r] = rs[r]; }
  }
}

// ---------------------------------------------------------------------------
// Kernel 3: max-over-heads arc weights + BCE; async K tiles across (s0, h).
// 12 heads per s-tile (even) => buffer parity is simply h&1.
// ---------------------------------------------------------------------------
__global__ void arc_bce_kernel(const f16* __restrict__ q_h,
                               const f16* __restrict__ k_h,
                               const float* __restrict__ amask,
                               const unsigned char* __restrict__ pad,
                               const float* __restrict__ rmax_g,
                               const float* __restrict__ rsum_g,
                               const int* __restrict__ trel,
                               float* __restrict__ accum) {
  const int t0 = blockIdx.x * 16, b = blockIdx.y;
  __shared__ alignas(32) f16 Qs[kH][16][kD];   // 24 KB
  __shared__ alignas(32) f16 Ks[2][16][kD];
  __shared__ alignas(16) int Ts[16][16];
  __shared__ float sm[kH][16], ss[kH][16];
  const int lane = threadIdx.x & 31, g = lane >> 4, m = lane & 15;
  for (int h = 0; h < kH; ++h) {
    const uint4* qsrc = (const uint4*)(q_h + (((size_t)b * kH + h) * kT + t0) * kD);
#pragma unroll
    for (int j = 0; j < 4; ++j) ((uint4*)Qs[h])[j * 32 + lane] = qsrc[j * 32 + lane];
    if (lane < 16) {
      size_t idx = ((size_t)b * kH + h) * kT + t0 + lane;
      sm[h][lane] = rmax_g[idx];
      ss[h][lane] = rsum_g[idx];
    }
  }
  auto issue = [&](int buf, int s0, int h) {
    const f16* src = k_h + (((size_t)b * kH + h) * kS + s0) * kD;
#pragma unroll
    for (int j = 0; j < 4; ++j) {
      int i = j * 32 + lane;
      cp16_async(&Ks[buf][0][0] + i * 8, src + i * 8);
    }
  };
  issue(0, 0, 0);
  __syncthreads();
  float acc = 0.f;
  for (int s0 = 0; s0 < kS; s0 += 16) {
    // stage target_rel tile [16 t][16 s] cooperatively (b128)
#pragma unroll
    for (int j = 0; j < 2; ++j) {
      int i = j * 32 + lane, r = i >> 2, q = i & 3;
      ((uint4*)Ts)[i] =
          *(const uint4*)(trel + ((size_t)(t0 + r) * kB + b) * kS + s0 + q * 4);
    }
    float am[8];
#pragma unroll
    for (int r = 0; r < 8; ++r) am[r] = 0.f;
    const bool pmask = pad[(size_t)b * kS + s0 + m] != 0;
    for (int h = 0; h < kH; ++h) {
      const int cur = h & 1;
      wait_async0();
      __syncthreads();
      if (h + 1 < kH) issue(cur ^ 1, s0, h + 1);
      else if (s0 + 16 < kS) issue(cur ^ 1, s0 + 16, 0);
      v8f c = {};
#pragma unroll
      for (int kc = 0; kc < kD; kc += 32) {
        v16h a = cat8(*(const v8h*)&Qs[h][m][kc + g * 8],
                      *(const v8h*)&Qs[h][m][kc + 16 + g * 8]);
        v16h bv = cat8(*(const v8h*)&Ks[cur][m][kc + g * 8], *(const v8h*)&Ks[cur][m][kc + 16 + g * 8]);
        c = wmma_f16(a, bv, c);
      }
#pragma unroll
      for (int r = 0; r < 8; ++r) {
        int M = 8 * g + r;
        float l = c[r] + amask[(size_t)(t0 + M) * kS + s0 + m];
        if (pmask) l = -INFINITY;
        float w = __expf(l - sm[h][M]) / ss[h][M];
        am[r] = fmaxf(am[r], w);
      }
      __syncthreads();
    }
#pragma unroll
    for (int r = 0; r < 8; ++r) {
      int M = 8 * g + r;
      int rel = Ts[M][m];
      float aw = am[r];
      float logp  = fmaxf(__logf(aw), -100.f);
      float log1m = fmaxf(log1pf(-aw), -100.f);
      float bce = (rel != 2) ? -logp : -log1m;
      if (rel == 0) bce = 0.f;
      acc += bce;
    }
    __syncthreads();   // protect Ts before next-tile staging
  }
#pragma unroll
  for (int off = 16; off > 0; off >>= 1) acc += __shfl_down(acc, off);
  if (lane == 0) atomicAdd(&accum[b], acc);
}

// ---------------------------------------------------------------------------
// Kernel 4: x = attn @ Wout.T + b_out -> d_out (f32); async A, staged stores.
// ---------------------------------------------------------------------------
__global__ void out_proj_kernel(const f16* __restrict__ attn_h,
                                const float* __restrict__ Wout,
                                const float* __restrict__ b_out,
                                float* __restrict__ x) {
  const int row0 = blockIdx.x * 16, col0 = blockIdx.y * 64;
  __shared__ alignas(32) f16 As[16][32];
  __shared__ alignas(32) f16 Bs[64][32];
  __shared__ alignas(16) float Csf[16][64];
  const int lane = threadIdx.x & 31, g = lane >> 4, m = lane & 15;
  v8f c[4] = {};
  for (int k0 = 0; k0 < kE; k0 += 32) {
    if (k0 + 32 < kE)
      __builtin_prefetch(Wout + (size_t)(col0 + 2 * lane) * kE + k0 + 32, 0, 1);
#pragma unroll
    for (int j = 0; j < 2; ++j) {                     // A (f16): async, b128
      int i = j * 32 + lane, r = i >> 2, q = i & 3;
      cp16_async(&As[r][q * 8], attn_h + (size_t)(row0 + r) * kE + k0 + q * 8);
    }
#pragma unroll
    for (int j = 0; j < 16; ++j) {                    // B^T: f32 -> f16
      int i = j * 32 + lane, n = i >> 3, q = i & 7;
      *(v4h*)&Bs[n][q * 4] = cvt4(((const float4*)(Wout + (size_t)(col0 + n) * kE + k0))[q]);
    }
    wait_async0();
    __syncthreads();
    v16h a = cat8(*(const v8h*)&As[m][g * 8], *(const v8h*)&As[m][16 + g * 8]);
#pragma unroll
    for (int nt = 0; nt < 4; ++nt) {
      v16h bv = cat8(*(const v8h*)&Bs[nt * 16 + m][g * 8], *(const v8h*)&Bs[nt * 16 + m][16 + g * 8]);
      c[nt] = wmma_f16(a, bv, c[nt]);
    }
    __syncthreads();
  }
#pragma unroll
  for (int nt = 0; nt < 4; ++nt) {
    float bias = b_out[col0 + nt * 16 + m];
#pragma unroll
    for (int r = 0; r < 8; ++r) Csf[8 * g + r][nt * 16 + m] = c[nt][r] + bias;
  }
  __syncthreads();
#pragma unroll
  for (int j = 0; j < 8; ++j) {
    int i = j * 32 + lane, r = i >> 4, q = i & 15;
    *(float4*)(x + (size_t)(row0 + r) * kE + col0 + q * 4) = ((float4*)Csf)[i];
  }
}

__global__ void init_accum_kernel(float* accum) {
  if (threadIdx.x < kB) accum[threadIdx.x] = 0.f;
}

__global__ void loss_final_kernel(const float* accum, const float* sid,
                                  float* arc_loss) {
  if (threadIdx.x < kB) arc_loss[threadIdx.x] = accum[threadIdx.x] * sid[threadIdx.x];
}

// ---------------------------------------------------------------------------
extern "C" void kernel_launch(void* const* d_in, const int* in_sizes, int n_in,
                              void* d_out, int out_size, void* d_ws, size_t ws_size,
                              hipStream_t stream) {
  (void)in_sizes; (void)n_in; (void)out_size; (void)ws_size;
  const float*         outs  = (const float*)d_in[2];
  const float*         gs    = (const float*)d_in[3];
  const unsigned char* pad   = (const unsigned char*)d_in[4];
  const float*         amask = (const float*)d_in[5];
  const float*         sid   = (const float*)d_in[6];
  const int*           trel  = (const int*)d_in[7];
  const float*         Win   = (const float*)d_in[8];
  const float*         b_in  = (const float*)d_in[9];
  const float*         Wout  = (const float*)d_in[10];
  const float*         b_out = (const float*)d_in[11];

  float* out      = (float*)d_out;
  float* arc_loss = out;
  float* outs_cpy = out + 8;
  float* x_out    = out + 8 + (size_t)kT * kB * kE;

  const size_t qkv_bytes  = (size_t)kB * kH * kT * kD * sizeof(f16);
  const size_t stat_bytes = (size_t)kB * kH * kT * sizeof(float);
  char* ws = (char*)d_ws;
  f16*   q_h    = (f16*)ws;                 ws += qkv_bytes;
  f16*   k_h    = (f16*)ws;                 ws += qkv_bytes;
  f16*   v_t    = (f16*)ws;                 ws += qkv_bytes;
  f16*   attn_h = (f16*)ws;                 ws += (size_t)kT * kB * kE * sizeof(f16);
  float* rmax_g = (float*)ws;               ws += stat_bytes;
  float* rsum_g = (float*)ws;               ws += stat_bytes;
  float* accum  = (float*)ws;               ws += 256;

  (void)hipMemcpyAsync(outs_cpy, outs, sizeof(float) * (size_t)kT * kB * kE,
                       hipMemcpyDeviceToDevice, stream);

  init_accum_kernel<<<1, 32, 0, stream>>>(accum);

  const dim3 pgrid((kT * kB) / 16, kE / 64);
  proj_kernel<0><<<pgrid, 32, 0, stream>>>(outs, Win, b_in, q_h);
  proj_kernel<1><<<pgrid, 32, 0, stream>>>(gs,   Win, b_in, k_h);
  proj_kernel<2><<<pgrid, 32, 0, stream>>>(gs,   Win, b_in, v_t);

  flash_attn_kernel<<<dim3(kT / 16, kH, kB), 32, 0, stream>>>(
      q_h, k_h, v_t, amask, pad, attn_h, rmax_g, rsum_g);

  arc_bce_kernel<<<dim3(kT / 16, kB), 32, 0, stream>>>(
      q_h, k_h, amask, pad, rmax_g, rsum_g, trel, accum);

  out_proj_kernel<<<pgrid, 32, 0, stream>>>(attn_h, Wout, b_out, x_out);

  loss_final_kernel<<<1, 32, 0, stream>>>(accum, sid, arc_loss);
}
